// PointNetSetAbstraction_70153995813097
// MI455X (gfx1250) — matrix operand order repaired and checked by reading, not compiled, not run on hardware
//
#include <hip/hip_runtime.h>
#include <hip/hip_bf16.h>
#include <stdint.h>

// ---------------- problem constants ----------------
#define B_   8
#define N_   8192
#define D_   16
#define S_   1024          // NPOINT
#define K_   32            // NSAMPLE
#define P_   (B_*S_*K_)    // 262144 total grouped points
#define R2_  0.04f         // RADIUS^2
#define EPS_ 1e-5f

typedef __attribute__((ext_vector_type(16))) _Float16 v16h;
typedef __attribute__((ext_vector_type(8)))  _Float16 v8h;
typedef __attribute__((ext_vector_type(8)))  float    v8f;
typedef __attribute__((ext_vector_type(4)))  uint32_t v4u;
typedef __attribute__((ext_vector_type(8)))  uint32_t v8u;

// ---------------- workspace layout (bytes) ----------------
static constexpr size_t NBLK     = (size_t)P_ / 16 / 8;           // 2048 gemm blocks
static constexpr size_t OFF_FPS  = 0;                              // B*S int32
static constexpr size_t OFF_W0H  = 32768;                          // 64x32 f16
static constexpr size_t OFF_W1H  = OFF_W0H + 64*32*2;              // 64x64 f16
static constexpr size_t OFF_W2H  = OFF_W1H + 64*64*2;              // 128x64 f16
static constexpr size_t OFF_COEF = OFF_W2H + 128*64*2;             // 3 * 256 f32 (a, bmix)
static constexpr size_t OFF_PART = 65536;                          // NBLK*2*128 f32 partial sums
static constexpr size_t OFF_X0   = OFF_PART + NBLK*2*128*4;        // (32,P) f16
static constexpr size_t OFF_Y0   = OFF_X0 + (size_t)32*P_*2;       // (64,P) f16
static constexpr size_t OFF_Y1   = OFF_Y0 + (size_t)64*P_*2;       // (64,P) f16
static constexpr size_t OFF_Y2   = OFF_Y1 + (size_t)64*P_*2;       // (128,P) f16

// ---------------- weight fp32 -> fp16 (with channel padding) ----------------
__global__ void cvt_w_kernel(const float* __restrict__ w, _Float16* __restrict__ wh,
                             int O, int Cin, int Cpad) {
  int i = blockIdx.x * blockDim.x + threadIdx.x;
  if (i >= O * Cpad) return;
  int c = i % Cpad, o = i / Cpad;
  wh[i] = (c < Cin) ? (_Float16)w[o * Cin + c] : (_Float16)0.f;
}

// ---------------- farthest point sampling ----------------
// one block per batch; 1024 threads (32 waves); each thread owns 8 points in registers.
__global__ __launch_bounds__(1024) void fps_kernel(const float* __restrict__ xyz,
                                                   const int* __restrict__ init_far,
                                                   int* __restrict__ fps_idx,
                                                   float* __restrict__ new_xyz) {
  __shared__ float redv[32];
  __shared__ int   redi[32];
  __shared__ int   sfar;
  const int b = blockIdx.x;
  const int tid = threadIdx.x;
  const int lane = tid & 31, wv = tid >> 5;
  const float* xb = xyz + (size_t)b * N_ * 3;

  float px[8], py[8], pz[8], dist[8];
#pragma unroll
  for (int j = 0; j < 8; ++j) {
    int p = tid + j * 1024;
    px[j] = xb[p * 3 + 0];
    py[j] = xb[p * 3 + 1];
    pz[j] = xb[p * 3 + 2];
    dist[j] = 1e10f;
  }
  int far = init_far[b];

  for (int it = 0; it < S_; ++it) {
    if (tid == 0) {
      fps_idx[b * S_ + it] = far;
      new_xyz[((size_t)b * S_ + it) * 3 + 0] = xb[far * 3 + 0];
      new_xyz[((size_t)b * S_ + it) * 3 + 1] = xb[far * 3 + 1];
      new_xyz[((size_t)b * S_ + it) * 3 + 2] = xb[far * 3 + 2];
    }
    const float cx = xb[far * 3 + 0], cy = xb[far * 3 + 1], cz = xb[far * 3 + 2];
    float bv = -1.f; int bp = 0x7fffffff;
#pragma unroll
    for (int j = 0; j < 8; ++j) {
      float dx = px[j] - cx, dy = py[j] - cy, dz = pz[j] - cz;
      float d = dx * dx + dy * dy + dz * dz;
      d = fminf(dist[j], d);
      dist[j] = d;
      int p = tid + j * 1024;
      if (d > bv || (d == bv && p < bp)) { bv = d; bp = p; }
    }
    // wave32 argmax (prefer smaller index on ties == jnp.argmax first-occurrence)
#pragma unroll
    for (int off = 16; off; off >>= 1) {
      float ov = __shfl_xor(bv, off, 32);
      int   op = __shfl_xor(bp, off, 32);
      if (ov > bv || (ov == bv && op < bp)) { bv = ov; bp = op; }
    }
    if (lane == 0) { redv[wv] = bv; redi[wv] = bp; }
    __syncthreads();
    if (tid < 32) {
      bv = redv[tid]; bp = redi[tid];
#pragma unroll
      for (int off = 16; off; off >>= 1) {
        float ov = __shfl_xor(bv, off, 32);
        int   op = __shfl_xor(bp, off, 32);
        if (ov > bv || (ov == bv && op < bp)) { bv = ov; bp = op; }
      }
      if (tid == 0) sfar = bp;
    }
    __syncthreads();
    far = sfar;
  }
}

// ---------------- ball query + grouping ----------------
__device__ inline void write_slot(_Float16* __restrict__ x0, int bs, int slot,
                                  const float* __restrict__ xb, const float* __restrict__ fb,
                                  int p, float cx, float cy, float cz) {
  const size_t pidx = (size_t)bs * K_ + slot;
  x0[(size_t)0 * P_ + pidx] = (_Float16)(xb[p * 3 + 0] - cx);
  x0[(size_t)1 * P_ + pidx] = (_Float16)(xb[p * 3 + 1] - cy);
  x0[(size_t)2 * P_ + pidx] = (_Float16)(xb[p * 3 + 2] - cz);
  const float* f = fb + (size_t)p * D_;
#pragma unroll
  for (int j = 0; j < D_; ++j) x0[(size_t)(3 + j) * P_ + pidx] = (_Float16)f[j];
#pragma unroll
  for (int c = 3 + D_; c < 32; ++c) x0[(size_t)c * P_ + pidx] = (_Float16)0.f;
}

// one wave per query point; select first K_ indices inside the ball, pad with first.
__global__ __launch_bounds__(256) void ball_group_kernel(const float* __restrict__ xyz,
                                                         const float* __restrict__ pts,
                                                         const float* __restrict__ new_xyz,
                                                         const int* __restrict__ fps_idx,
                                                         _Float16* __restrict__ x0) {
  const int lane = threadIdx.x & 31;
  const int bs = blockIdx.x * (blockDim.x >> 5) + (threadIdx.x >> 5);
  const int b = bs >> 10;
  const float cx = new_xyz[(size_t)bs * 3 + 0];
  const float cy = new_xyz[(size_t)bs * 3 + 1];
  const float cz = new_xyz[(size_t)bs * 3 + 2];
  const float* xb = xyz + (size_t)b * N_ * 3;
  const float* fb = pts + (size_t)b * N_ * D_;

  int cnt = 0;                     // uniform across wave
  int firstp = fps_idx[bs];        // fallback (center always qualifies numerically)
  for (int n0 = 0; n0 < N_ && cnt < K_; n0 += 32) {
    if (n0 + 32 < N_) __builtin_prefetch(&xb[(size_t)(n0 + 32 + lane) * 3], 0, 3);
    const int p = n0 + lane;
    const float dx = xb[p * 3 + 0] - cx;
    const float dy = xb[p * 3 + 1] - cy;
    const float dz = xb[p * 3 + 2] - cz;
    const bool in = (dx * dx + dy * dy + dz * dz) <= R2_;
    const uint32_t m = __builtin_amdgcn_ballot_w32(in);
    if (cnt == 0 && m) firstp = n0 + __builtin_ctz(m);
    const int pos = cnt + __popc(m & ((1u << lane) - 1u));
    if (in && pos < K_) write_slot(x0, bs, pos, xb, fb, p, cx, cy, cz);
    cnt += __popc(m);
    if (cnt > K_) cnt = K_;
  }
  for (int slot = cnt + lane; slot < K_; slot += 32)
    write_slot(x0, bs, slot, xb, fb, firstp, cx, cy, cz);
}

// ---------------- WMMA GEMM layer with TDM-staged activation tile ----------------
// activations channel-major (C, P) f16; weights (O, CIN) f16.
// Block stages its (CIN x 128) activation tile into LDS with one tensor_load_to_lds
// (TDM descriptor per CDNA5 ISA), then 8 waves each do a
// 16-point N-tile with v_wmma_f32_16x16x32_f16.
// A tile: lane m=lane&15, k0=8*(lane>=16); halves e<8 -> K=e+k0, e>=8 -> K=e+8+k0.
// B tile: lane&15 = N (point), element e = K (channel), same k order as A.
// D tile: element i -> O = mt*16 + i + 8*(lane>=16), N = lane&15.
template <int CIN, int COUT, bool BN_IN>
__global__ __launch_bounds__(256) void gemm_layer_kernel(const _Float16* __restrict__ xin,
                                                         const _Float16* __restrict__ wh,
                                                         const float* __restrict__ bias,
                                                         const float* __restrict__ coef,
                                                         _Float16* __restrict__ yout,
                                                         float* __restrict__ partials) {
  constexpr int KT = CIN / 32;
  constexpr int MT = COUT / 16;
  __shared__ __align__(64) _Float16 stile[CIN * 128];   // (CIN rows) x (128 points)
  __shared__ float sdsum[8][COUT];
  __shared__ float sdsq[8][COUT];

  const int lane = threadIdx.x & 31;
  const int wv = threadIdx.x >> 5;
  const int pblk = blockIdx.x * 128;          // block's 128-point slab
  const int pbase = pblk + wv * 16;           // this wave's 16-point N-tile

  // ---- TDM: DMA the (CIN x 128) f16 tile from (C,P) global layout into LDS ----
  if (wv == 0) {
    const uint64_t ga = (uint64_t)(uintptr_t)(xin + pblk);       // tile start VA
    const uint32_t lds = (uint32_t)(uintptr_t)(&stile[0]);       // LDS byte offset
    v4u g0;
    g0[0] = 1u;                                                  // count=1, user mode
    g0[1] = lds;                                                 // lds_addr
    g0[2] = (uint32_t)ga;                                        // global_addr[31:0]
    g0[3] = (uint32_t)((ga >> 32) & 0x01FFFFFFu) | (2u << 30);   // ga[56:32] | type=2
    v8u g1;
    g1[0] = 1u << 16;                                            // data_size=2B, mask=0
    g1[1] = ((uint32_t)P_ & 0xFFFFu) << 16;                      // tensor_dim0[15:0]
    g1[2] = (((uint32_t)P_ >> 16) & 0xFFFFu)                     // tensor_dim0[31:16]
          | (((uint32_t)CIN & 0xFFFFu) << 16);                   // tensor_dim1[15:0]
    g1[3] = (128u << 16);                                        // tile_dim0=128
    g1[4] = (uint32_t)CIN;                                       // tile_dim1=CIN, tile_dim2=0
    g1[5] = (uint32_t)P_;                                        // tensor_dim0_stride[31:0]
    g1[6] = 0u;                                                  // stride0[47:32]=0
    g1[7] = 0u;
    asm volatile("tensor_load_to_lds %0, %1" : : "s"(g0), "s"(g1) : "memory");
    __builtin_amdgcn_s_wait_tensorcnt(0);
  }
  __syncthreads();

  // ---- B fragments from LDS (optionally BN+ReLU-transformed) ----
  v16h bfrag[KT];
#pragma unroll
  for (int kt = 0; kt < KT; ++kt) {
    const int kb = kt * 32 + ((lane & 16) ? 8 : 0);
    v16h raw;
#pragma unroll
    for (int e = 0; e < 16; ++e) {
      const int c = kb + (e < 8 ? e : e + 8);
      raw[e] = stile[(size_t)c * 128 + wv * 16 + (lane & 15)];
      if constexpr (BN_IN) {
        const float a = coef[c], bm = coef[128 + c];
        raw[e] = (_Float16)fmaxf((float)raw[e] * a + bm, 0.f);
      }
    }
    bfrag[kt] = raw;
  }

  const int m = lane & 15;
  const int k0 = (lane & 16) ? 8 : 0;
  const int ohalf = (lane & 16) ? 8 : 0;

#pragma unroll
  for (int mt = 0; mt < MT; ++mt) {
    v8f acc = {};
#pragma unroll
    for (int kt = 0; kt < KT; ++kt) {
      const _Float16* wr = wh + (size_t)(mt * 16 + m) * CIN + kt * 32 + k0;
      v8h alo = *(const v8h*)wr;
      v8h ahi = *(const v8h*)(wr + 16);
      v16h afrag;
#pragma unroll
      for (int e = 0; e < 8; ++e) { afrag[e] = alo[e]; afrag[e + 8] = ahi[e]; }
      acc = __builtin_amdgcn_wmma_f32_16x16x32_f16(false, afrag, false, bfrag[kt],
                                                   (short)0, acc, false, false);
    }
#pragma unroll
    for (int i = 0; i < 8; ++i) {
      const int o = mt * 16 + i + ohalf;
      const float v = acc[i] + bias[o];
      yout[(size_t)o * P_ + pbase + m] = (_Float16)v;
      // deterministic width-16 shuffle reduction over N
      float s = v, q = v * v;
#pragma unroll
      for (int off = 8; off; off >>= 1) {
        s += __shfl_xor(s, off, 16);
        q += __shfl_xor(q, off, 16);
      }
      if (m == 0) { sdsum[wv][o] = s; sdsq[wv][o] = q; }
    }
  }
  __syncthreads();
  for (int o = threadIdx.x; o < COUT; o += 256) {
    float s = 0.f, q = 0.f;
#pragma unroll
    for (int w = 0; w < 8; ++w) { s += sdsum[w][o]; q += sdsq[w][o]; }
    partials[((size_t)blockIdx.x * 2 + 0) * 128 + o] = s;
    partials[((size_t)blockIdx.x * 2 + 1) * 128 + o] = q;
  }
}

// ---------------- BN finalize: coef[o]=g*rstd, coef[128+o]=be-mean*g*rstd ----------------
__global__ void finalize_kernel(const float* __restrict__ partials, int nblk,
                                const float* __restrict__ g, const float* __restrict__ be,
                                float* __restrict__ coef, int cout) {
  const int o = threadIdx.x;
  if (o >= cout) return;
  float s = 0.f, q = 0.f;
  for (int blk = 0; blk < nblk; ++blk) {         // fixed order -> deterministic
    s += partials[((size_t)blk * 2 + 0) * 128 + o];
    q += partials[((size_t)blk * 2 + 1) * 128 + o];
  }
  const float inv = 1.f / (float)P_;
  const float mean = s * inv;
  const float var = q * inv - mean * mean;
  const float rstd = rsqrtf(var + EPS_);
  const float a = g[o] * rstd;
  coef[o] = a;
  coef[128 + o] = be[o] - mean * a;
}

// ---------------- BN + ReLU + max over K, write (B,S,128) ----------------
__global__ __launch_bounds__(256) void final_max_kernel(const _Float16* __restrict__ y2,
                                                        const float* __restrict__ coef,
                                                        float* __restrict__ out) {
  const int gid = blockIdx.x * blockDim.x + threadIdx.x;
  if (gid >= B_ * S_ * 128) return;
  const int o = gid & 127;
  const int bs = gid >> 7;
  const float a = coef[o], bm = coef[128 + o];
  const _Float16* row = y2 + (size_t)o * P_ + (size_t)bs * K_;
  float mx = -3.4e38f;
#pragma unroll
  for (int k = 0; k < K_; ++k) mx = fmaxf(mx, (float)row[k] * a + bm);
  out[(size_t)bs * 128 + o] = fmaxf(mx, 0.f);   // relu commutes with max
}

// ---------------- launch ----------------
extern "C" void kernel_launch(void* const* d_in, const int* in_sizes, int n_in,
                              void* d_out, int out_size, void* d_ws, size_t ws_size,
                              hipStream_t stream) {
  (void)in_sizes; (void)n_in; (void)out_size; (void)ws_size;
  const float* xyz      = (const float*)d_in[0];
  const float* pts      = (const float*)d_in[1];
  const int*   init_far = (const int*)d_in[2];
  const float* w0 = (const float*)d_in[3];
  const float* b0 = (const float*)d_in[4];
  const float* g0 = (const float*)d_in[5];
  const float* be0 = (const float*)d_in[6];
  const float* w1 = (const float*)d_in[7];
  const float* b1 = (const float*)d_in[8];
  const float* g1 = (const float*)d_in[9];
  const float* be1 = (const float*)d_in[10];
  const float* w2 = (const float*)d_in[11];
  const float* b2 = (const float*)d_in[12];
  const float* g2 = (const float*)d_in[13];
  const float* be2 = (const float*)d_in[14];

  float* out = (float*)d_out;                       // new_xyz (B,S,3) then feats (B,S,128)
  char* ws = (char*)d_ws;
  int*       fps_idx = (int*)(ws + OFF_FPS);
  _Float16*  w0h = (_Float16*)(ws + OFF_W0H);
  _Float16*  w1h = (_Float16*)(ws + OFF_W1H);
  _Float16*  w2h = (_Float16*)(ws + OFF_W2H);
  float*     coef0 = (float*)(ws + OFF_COEF);
  float*     coef1 = coef0 + 256;
  float*     coef2 = coef0 + 512;
  float*     part = (float*)(ws + OFF_PART);
  _Float16*  x0 = (_Float16*)(ws + OFF_X0);
  _Float16*  y0 = (_Float16*)(ws + OFF_Y0);
  _Float16*  y1 = (_Float16*)(ws + OFF_Y1);
  _Float16*  y2 = (_Float16*)(ws + OFF_Y2);

  cvt_w_kernel<<<(64 * 32 + 255) / 256, 256, 0, stream>>>(w0, w0h, 64, 19, 32);
  cvt_w_kernel<<<(64 * 64 + 255) / 256, 256, 0, stream>>>(w1, w1h, 64, 64, 64);
  cvt_w_kernel<<<(128 * 64 + 255) / 256, 256, 0, stream>>>(w2, w2h, 128, 64, 64);

  fps_kernel<<<B_, 1024, 0, stream>>>(xyz, init_far, fps_idx, out);
  ball_group_kernel<<<(B_ * S_) / 8, 256, 0, stream>>>(xyz, pts, out, fps_idx, x0);

  gemm_layer_kernel<32, 64, false><<<(int)NBLK, 256, 0, stream>>>(x0, w0h, b0, coef0, y0, part);
  finalize_kernel<<<1, 128, 0, stream>>>(part, (int)NBLK, g0, be0, coef0, 64);
  gemm_layer_kernel<64, 64, true><<<(int)NBLK, 256, 0, stream>>>(y0, w1h, b1, coef0, y1, part);
  finalize_kernel<<<1, 128, 0, stream>>>(part, (int)NBLK, g1, be1, coef1, 64);
  gemm_layer_kernel<64, 128, true><<<(int)NBLK, 256, 0, stream>>>(y1, w2h, b2, coef1, y2, part);
  finalize_kernel<<<1, 128, 0, stream>>>(part, (int)NBLK, g2, be2, coef2, 128);

  final_max_kernel<<<(B_ * S_ * 128) / 256, 256, 0, stream>>>(y2, coef2, out + B_ * S_ * 3);
}
